// AttentionBlock_56581899157562
// MI455X (gfx1250) — matrix operand order repaired and checked by reading, not compiled, not run on hardware
//
#include <hip/hip_runtime.h>
#include <hip/hip_bf16.h>
#include <math.h>

// ---------------- problem constants ----------------
#define BB 4
#define CC 128
#define NN 4096
#define NGROUP 8
#define CPG 16          // channels per group
#define QSCALE 0.08838834764831845f   // 1/sqrt(128), folded into Q

typedef _Float16 v16h __attribute__((ext_vector_type(16)));
typedef float    v8f  __attribute__((ext_vector_type(8)));
typedef _Float16 h2v  __attribute__((ext_vector_type(2)));

// ---------------- WMMA helpers (gfx1250, wave32) ----------------
__device__ __forceinline__ v8f wmma16(v16h a, v16h b, v8f c) {
    // D = A(16x32 f16) * B(32x16 f16) + C(16x16 f32)
    return __builtin_amdgcn_wmma_f32_16x16x32_f16(false, a, false, b, (short)0, c, false, false);
}

// A fragment (16x32 f16): lane m = lane&15, k pairs at 2p+(p>=4?8:0)+(hi?8:0)
// rowp points at element [m][kbase] of a row-major (k-contiguous) matrix.
__device__ __forceinline__ v16h frag_a16(const _Float16* rowp, int hi) {
    v16h a;
#pragma unroll
    for (int p = 0; p < 8; ++p) {
        int k0 = 2 * p + ((p >= 4) ? 8 : 0) + (hi ? 8 : 0);
        h2v t = *(const h2v*)(rowp + k0);
        a[2 * p] = t[0]; a[2 * p + 1] = t[1];
    }
    return a;
}

// B fragment (32x16 f16): lane n = lane&15, k pairs at 2p+(p>=4?8:0)+(hi?8:0)
// colp points at element [n][kbase] of an n-major (k-contiguous) matrix.
__device__ __forceinline__ v16h frag_b16(const _Float16* colp, int hi) {
    v16h b;
#pragma unroll
    for (int p = 0; p < 8; ++p) {
        int k0 = 2 * p + ((p >= 4) ? 8 : 0) + (hi ? 8 : 0);
        h2v t = *(const h2v*)(colp + k0);
        b[2 * p] = t[0]; b[2 * p + 1] = t[1];
    }
    return b;
}

__device__ __forceinline__ float redmax16(float v) {
    v = fmaxf(v, __shfl_xor(v, 1, 32));
    v = fmaxf(v, __shfl_xor(v, 2, 32));
    v = fmaxf(v, __shfl_xor(v, 4, 32));
    v = fmaxf(v, __shfl_xor(v, 8, 32));
    return v;
}
__device__ __forceinline__ float redsum16(float v) {
    v += __shfl_xor(v, 1, 32);
    v += __shfl_xor(v, 2, 32);
    v += __shfl_xor(v, 4, 32);
    v += __shfl_xor(v, 8, 32);
    return v;
}

// ---- CDNA5 async global->LDS copy (ASYNCcnt path), 16B per lane ----------
// LDS address is taken from the vdst VGPR; flat addr low 32 bits give the LDS offset.
__device__ __forceinline__ void async_copy_b128(void* lds_ptr, const void* gptr) {
    unsigned lds_off = (unsigned)(unsigned long long)lds_ptr;
    asm volatile("global_load_async_to_lds_b128 %0, %1, off"
                 :: "v"(lds_off), "v"(gptr) : "memory");
}
__device__ __forceinline__ void wait_async0() {
    asm volatile("s_wait_asynccnt 0x0" ::: "memory");
}

// ---------------- kernel 1: GroupNorm statistics ----------------
// one block per (b, group); a group is a contiguous slab of 16*4096 floats
__global__ __launch_bounds__(256) void gn_stats_k(const float* __restrict__ x,
                                                  float* __restrict__ stats) {
    const int bg = blockIdx.x;                     // 0..31
    const float* base = x + (size_t)bg * (CPG * NN);
    float s = 0.f, ss = 0.f;
    for (int i = threadIdx.x; i < CPG * NN; i += 256) {
        float v = base[i];
        s += v; ss += v * v;
    }
    __shared__ float shs[256], shq[256];
    shs[threadIdx.x] = s; shq[threadIdx.x] = ss;
    __syncthreads();
    for (int off = 128; off > 0; off >>= 1) {
        if ((int)threadIdx.x < off) {
            shs[threadIdx.x] += shs[threadIdx.x + off];
            shq[threadIdx.x] += shq[threadIdx.x + off];
        }
        __syncthreads();
    }
    if (threadIdx.x == 0) {
        const float inv = 1.0f / (float)(CPG * NN);
        float mean = shs[0] * inv;
        float var  = shq[0] * inv - mean * mean;
        stats[2 * bg]     = mean;
        stats[2 * bg + 1] = rsqrtf(var + 1e-5f);
    }
}

// ---------------- kernel 2: convert weights to f16 ----------------
__global__ __launch_bounds__(256) void conv_w_k(const float* __restrict__ wqkv,
                                                const float* __restrict__ wout,
                                                _Float16* __restrict__ wqkv16,
                                                _Float16* __restrict__ wout16) {
    int idx = blockIdx.x * 256 + threadIdx.x;      // 65536 threads
    if (idx < 3 * CC * CC) wqkv16[idx] = (_Float16)wqkv[idx];
    else {
        int j = idx - 3 * CC * CC;
        if (j < CC * CC) wout16[j] = (_Float16)wout[j];
    }
}

// ---------------- kernel 3: apply GN, write h transposed [b][n][c] f16 ----
__global__ __launch_bounds__(256) void gn_apply_k(const float* __restrict__ x,
                                                  const float* __restrict__ gnw,
                                                  const float* __restrict__ gnb,
                                                  const float* __restrict__ stats,
                                                  _Float16* __restrict__ h_t) {
    int idx = blockIdx.x * 256 + threadIdx.x;      // B*C*N threads
    int b = idx >> 19;                             // C*N = 2^19
    int c = (idx >> 12) & (CC - 1);
    int n = idx & (NN - 1);
    int g = c >> 4;
    float mean = stats[2 * (b * NGROUP + g)];
    float rstd = stats[2 * (b * NGROUP + g) + 1];
    float h = (x[idx] - mean) * rstd * gnw[c] + gnb[c];
    h_t[((size_t)(b * NN + n)) * CC + c] = (_Float16)h;
}

// ---------------- kernel 4: QKV projection GEMM (WMMA) ----------------
// per batch: [384 x 128] x [128 x 4096]; block = (b, mb of 6, nb of 64),
// 4 waves, each wave: 16 output rows x 64 columns.
// Routing q/k/v is uniform in mb (scalar branch, no exec-mask churn).
__global__ __launch_bounds__(128) void qkv_gemm_k(const _Float16* __restrict__ wq16,
                                                  const float* __restrict__ bqkv,
                                                  const _Float16* __restrict__ h_t,
                                                  _Float16* __restrict__ q_t,
                                                  _Float16* __restrict__ k_t,
                                                  _Float16* __restrict__ v_m) {
    const int bid = blockIdx.x;
    const int nb   = bid & 63;
    const int rest = bid >> 6;
    const int mb   = rest % 6;
    const int b    = rest / 6;
    const int wv   = threadIdx.x >> 5;
    const int lane = threadIdx.x & 31;
    const int hi   = lane >> 4;
    const int ln   = lane & 15;

    const int o_row = mb * 64 + wv * 16 + ln;      // A-matrix row (m)
    v16h afr[4];
#pragma unroll
    for (int s = 0; s < 4; ++s)
        afr[s] = frag_a16(wq16 + (size_t)o_row * CC + 32 * s, hi);

    const int n_base = nb * 64;
#pragma unroll
    for (int jt = 0; jt < 4; ++jt) {
        const int n = n_base + jt * 16 + ln;       // B-matrix col per lane
        v8f acc = (v8f)0.0f;
#pragma unroll
        for (int s = 0; s < 4; ++s) {
            v16h bf = frag_b16(h_t + ((size_t)(b * NN + n)) * CC + 32 * s, hi);
            acc = wmma16(afr[s], bf, acc);
        }
        const int obase = mb * 64 + wv * 16 + 8 * hi;
        if (mb < 2) {                               // Q rows 0..127 (uniform)
#pragma unroll
            for (int p = 0; p < 8; ++p) {
                const int o = obase + p;
                q_t[((size_t)(b * NN + n)) * CC + o] =
                    (_Float16)((acc[p] + bqkv[o]) * QSCALE);
            }
        } else if (mb < 4) {                        // K rows 128..255 (uniform)
#pragma unroll
            for (int p = 0; p < 8; ++p) {
                const int o = obase + p;
                k_t[((size_t)(b * NN + n)) * CC + (o - CC)] =
                    (_Float16)(acc[p] + bqkv[o]);
            }
        } else {                                    // V rows 256..383 (uniform)
#pragma unroll
            for (int p = 0; p < 8; ++p) {
                const int o = obase + p;
                v_m[((size_t)(b * CC + (o - 2 * CC))) * NN + n] =
                    (_Float16)(acc[p] + bqkv[o]);
            }
        }
    }
}

// ---------------- kernel 5: flash attention (async double-buffered) ------
// block = (b, i-block of 64); 4 waves, wave owns 16 i rows.
// K tile [64 j][128 c] + V tile [128 c][64 j] ping-pong in LDS via
// global_load_async_to_lds_b128; P round-trips through per-wave LDS.
__global__ __launch_bounds__(128) void flash_attn_k(const _Float16* __restrict__ q_t,
                                                    const _Float16* __restrict__ k_t,
                                                    const _Float16* __restrict__ v_m,
                                                    _Float16* __restrict__ h2_t) {
    __shared__ _Float16 kt[2][64 * CC];    // [buf][j][c]  16KB each
    __shared__ _Float16 vt[2][CC * 64];    // [buf][c][j]  16KB each
    __shared__ _Float16 pt[4][16 * 64];    // per-wave P tile [i][j]

    const int bid = blockIdx.x;
    const int ib  = bid & 63;
    const int b   = bid >> 6;
    const int tid  = threadIdx.x;
    const int wv   = tid >> 5;
    const int lane = tid & 31;
    const int hi   = lane >> 4;
    const int ln   = lane & 15;

    const _Float16* ksrc = k_t + ((size_t)b * NN) * CC;   // [n][c] contiguous
    const _Float16* vsrc = v_m + ((size_t)b * CC) * NN;   // [c][n]

    // issue one tile's async copies: K = 8192 halfs contiguous, V = 128 rows x 64 halfs
    auto prefetch = [&](int buf, int j0) {
#pragma unroll
        for (int r = 0; r < 8; ++r) {
            int chunk = tid + 128 * r;                     // 1024 x 16B
            async_copy_b128(&kt[buf][chunk * 8], ksrc + (size_t)j0 * CC + chunk * 8);
        }
#pragma unroll
        for (int r = 0; r < 8; ++r) {                      // row c = tid
            async_copy_b128(&vt[buf][tid * 64 + r * 8],
                            vsrc + (size_t)tid * NN + j0 + r * 8);
        }
    };

    const int i_row = ib * 64 + wv * 16 + ln;
    v16h aq[4];
#pragma unroll
    for (int s = 0; s < 4; ++s)
        aq[s] = frag_a16(q_t + ((size_t)(b * NN + i_row)) * CC + 32 * s, hi);

    v8f oacc[8];
#pragma unroll
    for (int ct = 0; ct < 8; ++ct) oacc[ct] = (v8f)0.0f;
    float mrun[8], lrun[8];
#pragma unroll
    for (int p = 0; p < 8; ++p) { mrun[p] = -3.0e38f; lrun[p] = 0.f; }

    prefetch(0, 0);
    wait_async0();
    __syncthreads();

    int cur = 0;
    for (int j0 = 0; j0 < NN; j0 += 64) {
        if (j0 + 64 < NN) prefetch(cur ^ 1, j0 + 64);      // overlap with compute

        // ---- S = Q^T K (16 i x 64 j per wave)
        v8f sf[4];
#pragma unroll
        for (int jt = 0; jt < 4; ++jt) {
            v8f acc = (v8f)0.0f;
#pragma unroll
            for (int s = 0; s < 4; ++s) {
                v16h bf = frag_b16(&kt[cur][(jt * 16 + ln) * CC + 32 * s], hi);
                acc = wmma16(aq[s], bf, acc);
            }
            sf[jt] = acc;
        }

        // ---- online softmax (row m = p + 8*hi lives at VGPR p in each half)
        float mnew[8], sc[8], rs[8];
#pragma unroll
        for (int p = 0; p < 8; ++p) {
            float mx = fmaxf(fmaxf(sf[0][p], sf[1][p]), fmaxf(sf[2][p], sf[3][p]));
            mx = redmax16(mx);
            mnew[p] = fmaxf(mrun[p], mx);
            sc[p]   = __expf(mrun[p] - mnew[p]);
            rs[p]   = 0.f;
        }
#pragma unroll
        for (int jt = 0; jt < 4; ++jt) {
#pragma unroll
            for (int p = 0; p < 8; ++p) {
                float pv = __expf(sf[jt][p] - mnew[p]);
                rs[p] += pv;
                pt[wv][(p + 8 * hi) * 64 + jt * 16 + ln] = (_Float16)pv;
            }
        }
#pragma unroll
        for (int p = 0; p < 8; ++p) {
            lrun[p] = lrun[p] * sc[p] + redsum16(rs[p]);
            mrun[p] = mnew[p];
        }
        // rescale O accumulators
#pragma unroll
        for (int ct = 0; ct < 8; ++ct)
#pragma unroll
            for (int p = 0; p < 8; ++p) oacc[ct][p] *= sc[p];

        // ---- O += P * V^T  (M=i, N=c, K=j; A from own P tile, B from V tile)
#pragma unroll
        for (int ks = 0; ks < 2; ++ks) {
            v16h ap = frag_a16(&pt[wv][ln * 64 + 32 * ks], hi);
#pragma unroll
            for (int ct = 0; ct < 8; ++ct) {
                v16h bf = frag_b16(&vt[cur][(ct * 16 + ln) * 64 + 32 * ks], hi);
                oacc[ct] = wmma16(ap, bf, oacc[ct]);
            }
        }

        // all waves done reading `cur` AND prefetch into cur^1 landed
        wait_async0();
        __syncthreads();
        cur ^= 1;
    }

    // ---- normalize and store h2 transposed [b][i][c] f16
    float inv[8];
#pragma unroll
    for (int p = 0; p < 8; ++p) inv[p] = 1.0f / lrun[p];
#pragma unroll
    for (int ct = 0; ct < 8; ++ct) {
#pragma unroll
        for (int p = 0; p < 8; ++p) {
            const int ir = ib * 64 + wv * 16 + p + 8 * hi;
            h2_t[((size_t)(b * NN + ir)) * CC + ct * 16 + ln] =
                (_Float16)(oacc[ct][p] * inv[p]);
        }
    }
}

// ---------------- kernel 6: output projection + bias + residual ----------
__global__ __launch_bounds__(128) void out_gemm_k(const _Float16* __restrict__ wo16,
                                                  const float* __restrict__ bout,
                                                  const _Float16* __restrict__ h2_t,
                                                  const float* __restrict__ x,
                                                  float* __restrict__ out) {
    const int bid = blockIdx.x;
    const int nb   = bid & 63;
    const int rest = bid >> 6;
    const int mb   = rest & 1;
    const int b    = rest >> 1;
    const int wv   = threadIdx.x >> 5;
    const int lane = threadIdx.x & 31;
    const int hi   = lane >> 4;
    const int ln   = lane & 15;

    const int o_row = mb * 64 + wv * 16 + ln;
    v16h afr[4];
#pragma unroll
    for (int s = 0; s < 4; ++s)
        afr[s] = frag_a16(wo16 + (size_t)o_row * CC + 32 * s, hi);

#pragma unroll
    for (int jt = 0; jt < 4; ++jt) {
        const int n = nb * 64 + jt * 16 + ln;
        v8f acc = (v8f)0.0f;
#pragma unroll
        for (int s = 0; s < 4; ++s) {
            v16h bf = frag_b16(h2_t + ((size_t)(b * NN + n)) * CC + 32 * s, hi);
            acc = wmma16(afr[s], bf, acc);
        }
#pragma unroll
        for (int p = 0; p < 8; ++p) {
            const int o = mb * 64 + wv * 16 + p + 8 * hi;
            const size_t oi = ((size_t)(b * CC + o)) * NN + n;
            out[oi] = acc[p] + bout[o] + x[oi];
        }
    }
}

// ---------------- host launcher ----------------
extern "C" void kernel_launch(void* const* d_in, const int* in_sizes, int n_in,
                              void* d_out, int out_size, void* d_ws, size_t ws_size,
                              hipStream_t stream) {
    (void)in_sizes; (void)n_in; (void)out_size; (void)ws_size;
    const float* x    = (const float*)d_in[0];
    const float* gnw  = (const float*)d_in[1];
    const float* gnb  = (const float*)d_in[2];
    const float* wqkv = (const float*)d_in[3];
    const float* bqkv = (const float*)d_in[4];
    const float* wout = (const float*)d_in[5];
    const float* bout = (const float*)d_in[6];
    float* out = (float*)d_out;

    char* ws = (char*)d_ws;
    // workspace layout (256B-aligned offsets)
    float*    stats  = (float*)(ws + 0);                       // 32 * 2 f32
    _Float16* wqkv16 = (_Float16*)(ws + 256);                  // 49152 f16
    _Float16* wout16 = (_Float16*)(ws + 98560);                // 16384 f16
    _Float16* h_t    = (_Float16*)(ws + 131328);               // B*N*C f16
    _Float16* q_t    = (_Float16*)(ws + 131328 + 1 * 4194304); // B*N*C f16
    _Float16* k_t    = (_Float16*)(ws + 131328 + 2 * 4194304);
    _Float16* v_m    = (_Float16*)(ws + 131328 + 3 * 4194304);
    _Float16* h2_t   = (_Float16*)(ws + 131328 + 4 * 4194304);

    gn_stats_k<<<BB * NGROUP, 256, 0, stream>>>(x, stats);
    conv_w_k<<<(3 * CC * CC + CC * CC + 255) / 256, 256, 0, stream>>>(wqkv, wout, wqkv16, wout16);
    gn_apply_k<<<(BB * CC * NN) / 256, 256, 0, stream>>>(x, gnw, gnb, stats, h_t);
    qkv_gemm_k<<<BB * 6 * (NN / 64), 128, 0, stream>>>(wqkv16, bqkv, h_t, q_t, k_t, v_m);
    flash_attn_k<<<BB * (NN / 64), 128, 0, stream>>>(q_t, k_t, v_m, h2_t);
    out_gemm_k<<<BB * 2 * (NN / 64), 128, 0, stream>>>(wout16, bout, h2_t, x, out);
}
